// GAT_24833500905997
// MI455X (gfx1250) — compile-verified, matrix operand order checked by reading, not run on hardware
//
#include <hip/hip_runtime.h>
#include <hip/hip_bf16.h>
#include <math.h>

// ---------------------------------------------------------------------------
// Problem constants (from reference): N=8192 nodes, E=262144 edges,
// F1=256, F2=128.  E' = E + N self loops.
// ---------------------------------------------------------------------------
#define NN   8192
#define EE   262144
#define EP   270336          // EE + NN
#define F1   256
#define F2   128
#define SLOPE 0.2f

typedef __attribute__((ext_vector_type(16))) _Float16 v16h;
typedef __attribute__((ext_vector_type(8)))  float    v8f;

// ---------------------------------------------------------------------------
// Fragment loaders.  CDNA5 wave32 WMMA 16x16x32 f16 layouts (ISA 7.12.2):
//   A (16x32, MxK): lane l holds row M = l&15; half h holds
//                   K = (l>>4)*8 + (h&7) + (h>>3)*16
//   B (32x16, KxN): lane l holds column N = l&15 with the same K mapping.
//   C/D (16x16 f32): element [m][n] in vgpr v of lane l with
//                   m = v + 8*(l>>4), n = l&15
// ---------------------------------------------------------------------------
__device__ __forceinline__ v16h load_frag_rowmajor_f16(
    const _Float16* __restrict__ base, int stride, int row, int k0, int lane) {
  const _Float16* p = base + (size_t)row * stride + k0 + ((lane >> 4) << 3);
  v16h r;
#pragma unroll
  for (int h = 0; h < 8; ++h) r[h] = p[h];
#pragma unroll
  for (int h = 0; h < 8; ++h) r[h + 8] = p[16 + h];
  return r;
}

__device__ __forceinline__ v16h load_frag_packed(
    const _Float16* __restrict__ fragbuf, int fragIdx, int lane) {
  const _Float16* p = fragbuf + (size_t)fragIdx * 512 + lane * 16;
  v16h r;
#pragma unroll
  for (int h = 0; h < 16; ++h) r[h] = p[h];
  return r;
}

// Pre-pack a row-major f32 [K][Nc] weight matrix into per-fragment-contiguous
// f16 layout: frag (nt, kk) -> 512 halves, lane-major, matching the B layout.
__global__ __launch_bounds__(256) void k_cvt_bfrag(
    const float* __restrict__ src, _Float16* __restrict__ dst, int K, int Nc) {
  int t = blockIdx.x * blockDim.x + threadIdx.x;
  int KS = K >> 5;
  int nfrag = (Nc >> 4) * KS;
  if (t >= nfrag * 32) return;
  int frag = t >> 5, lane = t & 31;
  int nt = frag / KS, kk = frag % KS;
  _Float16* p = dst + (size_t)frag * 512 + lane * 16;
  int col = nt * 16 + (lane & 15);
  int kb  = kk * 32 + ((lane >> 4) << 3);
#pragma unroll
  for (int h = 0; h < 16; ++h) {
    int k = kb + (h & 7) + ((h >> 3) << 4);
    p[h] = (_Float16)src[(size_t)k * Nc + col];
  }
}

// ---------------------------------------------------------------------------
// GEMM1: h1 = x @ W1   (M=8192, K=8192, N=256).
// x (fp32) is staged into LDS with the CDNA5 async global->LDS DMA path
// (global_load_async_to_lds_b128 / ASYNCcnt), double-buffered: the copy of
// the (kk+1) 16x64 tile overlaps the two WMMAs of tile kk.  The 8 waves per
// block (8 N-tiles) all consume the same x tile, so each x element crosses
// HBM once per grid.y slice.  grid = (512, 2), block = 256.
// ---------------------------------------------------------------------------
__global__ __launch_bounds__(256) void k_gemm1(
    const float* __restrict__ x, const _Float16* __restrict__ w1frag,
    float* __restrict__ h1) {
  __shared__ __align__(16) float sh[2][16 * 64];
  const int tid  = threadIdx.x;
  const int lane = tid & 31;
  const int wave = tid >> 5;
  const int mt = blockIdx.x;
  const int nt = blockIdx.y * 8 + wave;   // 0..15
  const int row = lane & 15;
  const int kb  = (lane >> 4) << 3;

  // cooperative tile copy: thread -> one 16-byte (4-float) chunk of the
  // 16x64 tile.  r = tid>>4 (row), c = (tid&15)*4 (col).
  const int cr = tid >> 4;
  const int cc = (tid & 15) << 2;
  const float* gsrc = x + (size_t)(mt * 16 + cr) * NN + cc;
  const unsigned lds0 = (unsigned)(size_t)(&sh[0][cr * 64 + cc]);
  const unsigned lds1 = (unsigned)(size_t)(&sh[1][cr * 64 + cc]);

  v8f acc = {};
  const int KS2 = NN >> 6;                // 128 iterations of K=64

  // prologue: async-copy tile 0 into buffer 0
  {
    unsigned long long ga = (unsigned long long)(const void*)gsrc;
    asm volatile("global_load_async_to_lds_b128 %0, %1, off"
                 :: "v"(lds0), "v"(ga) : "memory");
  }
  for (int kk = 0; kk < KS2; ++kk) {
    __syncthreads();                      // prev compute done: next buf reusable
    if (kk + 1 < KS2) {
      unsigned dst = ((kk + 1) & 1) ? lds1 : lds0;
      unsigned long long ga =
          (unsigned long long)(const void*)(gsrc + (size_t)(kk + 1) * 64);
      asm volatile("global_load_async_to_lds_b128 %0, %1, off"
                   :: "v"(dst), "v"(ga) : "memory");
      asm volatile("s_wait_asynccnt 0x1" ::: "memory");   // tile kk landed
    } else {
      asm volatile("s_wait_asynccnt 0x0" ::: "memory");
    }
    __syncthreads();                      // tile kk visible to all waves
    const float* s = sh[kk & 1];
#pragma unroll
    for (int sub = 0; sub < 2; ++sub) {
      v16h a;
      const float* sp = s + row * 64 + sub * 32 + kb;
#pragma unroll
      for (int h = 0; h < 8; ++h) a[h]     = (_Float16)sp[h];
#pragma unroll
      for (int h = 0; h < 8; ++h) a[h + 8] = (_Float16)sp[16 + h];
      v16h b = load_frag_packed(w1frag, nt * (NN >> 5) + kk * 2 + sub, lane);
      acc = __builtin_amdgcn_wmma_f32_16x16x32_f16(
          false, a, false, b, (short)0, acc, false, false);
    }
  }
  const int m0 = mt * 16, n0 = nt * 16, roff = (lane >> 4) << 3, n = lane & 15;
#pragma unroll
  for (int v = 0; v < 8; ++v)
    h1[(size_t)(m0 + v + roff) * F1 + n0 + n] = acc[v];
}

// ---------------------------------------------------------------------------
// Generic GEMM: C[M x Nc] = A_f16[M x K] @ Bfrag  (A row-major f16, B packed)
// one 16x16 tile per wave; grid.x = M/16, waves cover N tiles.
// ---------------------------------------------------------------------------
__global__ __launch_bounds__(256) void k_gemm_af16(
    const _Float16* __restrict__ A, const _Float16* __restrict__ bfrag,
    float* __restrict__ C, int K, int Nc) {
  const int tid = threadIdx.x, lane = tid & 31, wave = tid >> 5;
  const int mt = blockIdx.x;
  const int nt = blockIdx.y * (blockDim.x >> 5) + wave;
  if (nt * 16 >= Nc) return;
  const int KS = K >> 5;
  v8f acc = {};
  for (int kk = 0; kk < KS; ++kk) {
    v16h a = load_frag_rowmajor_f16(A, K, mt * 16 + (lane & 15), kk * 32, lane);
    v16h b = load_frag_packed(bfrag, nt * KS + kk, lane);
    acc = __builtin_amdgcn_wmma_f32_16x16x32_f16(
        false, a, false, b, (short)0, acc, false, false);
  }
  const int m0 = mt * 16, n0 = nt * 16, roff = (lane >> 4) << 3, n = lane & 15;
#pragma unroll
  for (int v = 0; v < 8; ++v)
    C[(size_t)(m0 + v + roff) * Nc + n0 + n] = acc[v];
}

// ---------------------------------------------------------------------------
// GEMM3: adj = sigmoid(z @ z^T)  (M=N=8192, K=128).  B[k][n] = z[n][k], so the
// B fragment is just another row-major read of z_f16 (2 MB -> L2 resident).
// Output is a 256 MB stream written exactly once -> non-temporal stores.
// grid = (512, 64).
// ---------------------------------------------------------------------------
__global__ __launch_bounds__(256) void k_gemm3(
    const _Float16* __restrict__ zf, float* __restrict__ adj) {
  const int tid = threadIdx.x, lane = tid & 31, wave = tid >> 5;
  const int mt = blockIdx.x;
  const int nt = blockIdx.y * 8 + wave;
  v8f acc = {};
#pragma unroll
  for (int kk = 0; kk < F2 / 32; ++kk) {
    v16h a = load_frag_rowmajor_f16(zf, F2, mt * 16 + (lane & 15), kk * 32, lane);
    v16h b = load_frag_rowmajor_f16(zf, F2, nt * 16 + (lane & 15), kk * 32, lane);
    acc = __builtin_amdgcn_wmma_f32_16x16x32_f16(
        false, a, false, b, (short)0, acc, false, false);
  }
  const int m0 = mt * 16, n0 = nt * 16, roff = (lane >> 4) << 3, n = lane & 15;
#pragma unroll
  for (int v = 0; v < 8; ++v) {
    float s = 1.0f / (1.0f + __expf(-acc[v]));
    __builtin_nontemporal_store(s, &adj[(size_t)(m0 + v + roff) * NN + n0 + n]);
  }
}

// ---------------------------------------------------------------------------
// Attention score GEMVs: alpha_s[i] = h[i,:].a_src ; alpha_d[i] = h[i,:].a_dst
// one wave32 per node.
// ---------------------------------------------------------------------------
__global__ __launch_bounds__(256) void k_alpha(
    const float* __restrict__ h, const float* __restrict__ asrc,
    const float* __restrict__ adst, float* __restrict__ out_s,
    float* __restrict__ out_d, int F) {
  int node = blockIdx.x * (blockDim.x >> 5) + (threadIdx.x >> 5);
  int lane = threadIdx.x & 31;
  if (node >= NN) return;
  float ss = 0.f, sd = 0.f;
  for (int c = lane; c < F; c += 32) {
    float v = h[(size_t)node * F + c];
    ss += v * asrc[c];
    sd += v * adst[c];
  }
#pragma unroll
  for (int off = 16; off; off >>= 1) {
    ss += __shfl_xor(ss, off);
    sd += __shfl_xor(sd, off);
  }
  if (lane == 0) { out_s[node] = ss; out_d[node] = sd; }
}

// ---------------------------------------------------------------------------
// Edge kernels (segment softmax + weighted aggregation)
// ---------------------------------------------------------------------------
__device__ __forceinline__ void atomicMaxF(float* addr, float val) {
  if (val >= 0.f) atomicMax((int*)addr, __float_as_int(val));
  else            atomicMin((unsigned int*)addr, __float_as_uint(val));
}

__global__ __launch_bounds__(256) void k_fill(float* __restrict__ p, int n, float v) {
  int t = blockIdx.x * blockDim.x + threadIdx.x;
  if (t < n) p[t] = v;
}

__global__ __launch_bounds__(256) void k_edge_max(
    const int* __restrict__ eidx, const float* __restrict__ as,
    const float* __restrict__ ad, float* __restrict__ ebuf,
    float* __restrict__ segmax) {
  int t = blockIdx.x * blockDim.x + threadIdx.x;
  if (t >= EP) return;
  int s = (t < EE) ? eidx[t]      : t - EE;
  int d = (t < EE) ? eidx[EE + t] : t - EE;
  float v = as[s] + ad[d];
  v = (v > 0.f) ? v : SLOPE * v;            // leaky_relu
  ebuf[t] = v;
  atomicMaxF(&segmax[d], v);
}

__global__ __launch_bounds__(256) void k_edge_sum(
    const int* __restrict__ eidx, const float* __restrict__ ebuf,
    const float* __restrict__ segmax, float* __restrict__ wbuf,
    float* __restrict__ segsum) {
  int t = blockIdx.x * blockDim.x + threadIdx.x;
  if (t >= EP) return;
  int d = (t < EE) ? eidx[EE + t] : t - EE;
  float w = __expf(ebuf[t] - segmax[d]);
  wbuf[t] = w;
  atomicAdd(&segsum[d], w);
}

__global__ __launch_bounds__(256) void k_edge_agg(
    const int* __restrict__ eidx, const float* __restrict__ h,
    const float* __restrict__ wbuf, const float* __restrict__ segsum,
    float* __restrict__ agg, int F) {
  int t = blockIdx.x * blockDim.x + threadIdx.x;
  int C4 = F >> 2;
  int e = t / C4, c = (t % C4) << 2;
  if (e >= EP) return;
  int s = (e < EE) ? eidx[e]      : e - EE;
  int d = (e < EE) ? eidx[EE + e] : e - EE;
  float alpha = wbuf[e] / (segsum[d] + 1e-16f);
  const float4 hv = *(const float4*)(h + (size_t)s * F + c);
  float* ap = agg + (size_t)d * F + c;
  atomicAdd(ap + 0, alpha * hv.x);
  atomicAdd(ap + 1, alpha * hv.y);
  atomicAdd(ap + 2, alpha * hv.z);
  atomicAdd(ap + 3, alpha * hv.w);
}

// ---------------------------------------------------------------------------
// Post-processing: bias + relu -> f16 (layer1);  bias + tanh -> z (layer2)
// ---------------------------------------------------------------------------
__global__ __launch_bounds__(256) void k_post1(
    const float* __restrict__ agg, const float* __restrict__ bias,
    _Float16* __restrict__ outh, int F, int total) {
  int t = blockIdx.x * blockDim.x + threadIdx.x;
  if (t >= total) return;
  float v = agg[t] + bias[t & (F - 1)];
  v = fmaxf(v, 0.f);
  outh[t] = (_Float16)v;
}

__global__ __launch_bounds__(256) void k_post2(
    const float* __restrict__ agg, const float* __restrict__ bias,
    float* __restrict__ zout, _Float16* __restrict__ zh, int F, int total) {
  int t = blockIdx.x * blockDim.x + threadIdx.x;
  if (t >= total) return;
  float v = tanhf(agg[t] + bias[t & (F - 1)]);
  zout[t] = v;
  zh[t] = (_Float16)v;
}

// ---------------------------------------------------------------------------
// Host launcher
// ---------------------------------------------------------------------------
extern "C" void kernel_launch(void* const* d_in, const int* in_sizes, int n_in,
                              void* d_out, int out_size, void* d_ws, size_t ws_size,
                              hipStream_t stream) {
  const float* x   = (const float*)d_in[0];
  const int*   eid = (const int*)  d_in[1];
  const float* W1  = (const float*)d_in[2];
  const float* as1 = (const float*)d_in[3];
  const float* ad1 = (const float*)d_in[4];
  const float* b1  = (const float*)d_in[5];
  const float* W2  = (const float*)d_in[6];
  const float* as2 = (const float*)d_in[7];
  const float* ad2 = (const float*)d_in[8];
  const float* b2  = (const float*)d_in[9];
  float* out = (float*)d_out;

  // ws layout (~13 MB)
  char* ws = (char*)d_ws;
  _Float16* w1frag = (_Float16*)(ws + 0);          // 4 MB
  _Float16* h1rel  = (_Float16*)(ws + 4194304);    // 4 MB
  _Float16* w2frag = (_Float16*)(ws + 8388608);    // 64 KB
  _Float16* zf     = (_Float16*)(ws + 8454144);    // 2 MB
  float* al_s   = (float*)(ws + 10551296);         // 32 KB
  float* al_d   = (float*)(ws + 10584064);         // 32 KB
  float* segmax = (float*)(ws + 10616832);         // 32 KB
  float* segsum = (float*)(ws + 10649600);         // 32 KB
  float* ebuf   = (float*)(ws + 10682368);         // ~1.03 MB
  float* wbuf   = (float*)(ws + 11763712);         // ~1.03 MB

  // large f32 intermediates live inside the adj region of d_out; adj is the
  // last thing written (GEMM3 reads only zf from ws, so no overlap hazard).
  float* h1   = out + 0;          // 8192*256
  float* agg1 = out + 2097152;    // 8192*256
  float* h2   = out + 4194304;    // 8192*128
  float* agg2 = out + 5242880;    // 8192*128
  float* zout = out + 67108864;   // z output region (8192*128)

  // ---- Layer 1 ----
  k_cvt_bfrag<<<512, 256, 0, stream>>>(W1, w1frag, NN, F1);
  k_gemm1<<<dim3(512, 2), 256, 0, stream>>>(x, w1frag, h1);
  k_alpha<<<1024, 256, 0, stream>>>(h1, as1, ad1, al_s, al_d, F1);
  k_fill<<<32, 256, 0, stream>>>(segmax, NN, -INFINITY);
  k_fill<<<32, 256, 0, stream>>>(segsum, NN, 0.f);
  k_fill<<<8192, 256, 0, stream>>>(agg1, NN * F1, 0.f);
  k_edge_max<<<1056, 256, 0, stream>>>(eid, al_s, al_d, ebuf, segmax);
  k_edge_sum<<<1056, 256, 0, stream>>>(eid, ebuf, segmax, wbuf, segsum);
  k_edge_agg<<<67584, 256, 0, stream>>>(eid, h1, wbuf, segsum, agg1, F1);
  k_post1<<<8192, 256, 0, stream>>>(agg1, b1, h1rel, F1, NN * F1);

  // ---- Layer 2 ----
  k_cvt_bfrag<<<8, 256, 0, stream>>>(W2, w2frag, F1, F2);
  k_gemm_af16<<<dim3(512, 1), 256, 0, stream>>>(h1rel, w2frag, h2, F1, F2);
  k_alpha<<<1024, 256, 0, stream>>>(h2, as2, ad2, al_s, al_d, F2);
  k_fill<<<32, 256, 0, stream>>>(segmax, NN, -INFINITY);
  k_fill<<<32, 256, 0, stream>>>(segsum, NN, 0.f);
  k_fill<<<4096, 256, 0, stream>>>(agg2, NN * F2, 0.f);
  k_edge_max<<<1056, 256, 0, stream>>>(eid, al_s, al_d, ebuf, segmax);
  k_edge_sum<<<1056, 256, 0, stream>>>(eid, ebuf, segmax, wbuf, segsum);
  k_edge_agg<<<33792, 256, 0, stream>>>(eid, h2, wbuf, segsum, agg2, F2);
  k_post2<<<4096, 256, 0, stream>>>(agg2, b2, zout, zf, F2, NN * F2);

  // ---- Decoder: adj = sigmoid(z z^T) ----
  k_gemm3<<<dim3(512, 64), 256, 0, stream>>>(zf, out);
}
